// Seq2SeqModel_25752623906857
// MI455X (gfx1250) — matrix operand order repaired and checked by reading, not compiled, not run on hardware
//
#include <hip/hip_runtime.h>
#include <stdint.h>

// Seq2Seq GRU (B=128, T=256, F=64, H=1024, P=512, OUT=48) for gfx1250.
// Strategy: bf16 WMMA (v_wmma_f32_16x16x32_bf16) for every GEMM, f32
// accumulation, fused GRU gate epilogue, double-buffered hidden state,
// one kernel launch per sequential stage (graph-captured by harness).

#define Bsz  128
#define Tlen 256
#define Feat 64
#define Hid  1024
#define Prj  512
#define OUTL 48

typedef __attribute__((ext_vector_type(8)))  unsigned short ushort8;
typedef __attribute__((ext_vector_type(16))) unsigned short ushort16;
typedef __attribute__((ext_vector_type(16))) __bf16        v16bf;
typedef __attribute__((ext_vector_type(8)))  float         v8f;

static __device__ __forceinline__ unsigned short f32_to_bf16(float f){
  unsigned int u = __float_as_uint(f);
  u += 0x7fffu + ((u >> 16) & 1u);          // round-to-nearest-even
  return (unsigned short)(u >> 16);
}

static __device__ __forceinline__ float sigm(float x){
  x = fminf(fmaxf(x, -30.0f), 30.0f);
  return 1.0f / (1.0f + __expf(-x));
}
static __device__ __forceinline__ float tanh_fast(float x){
  x = fminf(fmaxf(x, -15.0f), 15.0f);
  float e = __expf(2.0f * x);
  return (e - 1.0f) / (e + 1.0f);
}

// A fragment (16x32 bf16, row-major source with row stride lda):
// lane<16: row M=lane,   K = k0+0..7  (v0..3) and k0+16..23 (v4..7)
// lane>=16: row M=lane-16, K = k0+8..15 and k0+24..31
static __device__ __forceinline__ v16bf load_a_frag(const unsigned short* A, int lda,
                                                    int m0, int k0, int lane){
  const int hi = lane >> 4;
  const int m  = m0 + (lane & 15);
  const unsigned short* p = A + (size_t)m * (size_t)lda + (size_t)(k0 + hi * 8);
  ushort8 c0 = *(const ushort8*)(p);
  ushort8 c1 = *(const ushort8*)(p + 16);
  union { ushort16 u; v16bf b; } t;
#pragma unroll
  for (int i = 0; i < 8; ++i) { t.u[i] = c0[i]; t.u[i + 8] = c1[i]; }
  return t.b;
}

// B fragment (32x16 bf16) from K-major packed weight W[k*ldw + n]:
// lane l holds col n0+(l&15); lane<16: K = k0+0..7, k0+16..23; lane>=16: K = k0+8..15, k0+24..31.
static __device__ __forceinline__ v16bf load_b_frag(const unsigned short* W, int ldw,
                                                    int k0, int n0, int lane){
  const int hi = lane >> 4;
  const unsigned short* p = W + (size_t)(k0 + hi * 8) * (size_t)ldw + (size_t)(n0 + (lane & 15));
  union { ushort16 u; v16bf b; } t;
#pragma unroll
  for (int i = 0; i < 8; ++i) { t.u[i] = p[(size_t)i * ldw]; t.u[i + 8] = p[(size_t)(i + 16) * ldw]; }
  return t.b;
}

static __device__ __forceinline__ v8f wmma_bf16(v16bf a, v16bf b, v8f c){
  return __builtin_amdgcn_wmma_f32_16x16x32_bf16(false, a, false, b, (short)0, c, false, false);
}

// One GRU layer-step: h_new = GRU(x, h_prev). Block = 4 waves x 16 rows,
// grid = (H/64 column blocks, 2 row blocks). Each wave computes a 16x64
// tile for all three gates (r,z shared x+h accumulators; n kept separate).
__global__ __launch_bounds__(128) void gru_stage(
    const unsigned short* __restrict__ Ax, int ldax, int Kx,
    const unsigned short* __restrict__ Wx,   // [Kx, 3H] K-major bf16
    const unsigned short* __restrict__ Ah,   // [B, H] bf16, lda = H
    const unsigned short* __restrict__ Wh,   // [H, 3H] K-major bf16
    const float* __restrict__ bih, const float* __restrict__ bhh,
    const float* __restrict__ h_prev,
    float* __restrict__ h_new, unsigned short* __restrict__ h_new_bf)
{
  const int lane = threadIdx.x & 31;
  const int wave = threadIdx.x >> 5;
  const int m0 = blockIdx.y * 64 + wave * 16;
  const int c0 = blockIdx.x * 64;

  v8f accR[4] = {}, accZ[4] = {}, accIN[4] = {}, accHN[4] = {};

  for (int k0 = 0; k0 < Kx; k0 += 32){
    v16bf a = load_a_frag(Ax, ldax, m0, k0, lane);
#pragma unroll
    for (int j = 0; j < 4; ++j){
      const int n = c0 + 16 * j;
      accR[j]  = wmma_bf16(a, load_b_frag(Wx, 3*Hid, k0,           n, lane), accR[j]);
      accZ[j]  = wmma_bf16(a, load_b_frag(Wx, 3*Hid, k0,   Hid + n, lane), accZ[j]);
      accIN[j] = wmma_bf16(a, load_b_frag(Wx, 3*Hid, k0, 2*Hid + n, lane), accIN[j]);
    }
  }
  for (int k0 = 0; k0 < Hid; k0 += 32){
    v16bf a = load_a_frag(Ah, Hid, m0, k0, lane);
#pragma unroll
    for (int j = 0; j < 4; ++j){
      const int n = c0 + 16 * j;
      accR[j]  = wmma_bf16(a, load_b_frag(Wh, 3*Hid, k0,           n, lane), accR[j]);
      accZ[j]  = wmma_bf16(a, load_b_frag(Wh, 3*Hid, k0,   Hid + n, lane), accZ[j]);
      accHN[j] = wmma_bf16(a, load_b_frag(Wh, 3*Hid, k0, 2*Hid + n, lane), accHN[j]);
    }
  }

  // C/D layout: VGPR v, lanes 0-15 -> row m0+v, lanes 16-31 -> row m0+8+v.
  const int cl   = lane & 15;
  const int rofs = (lane >> 4) * 8;
#pragma unroll
  for (int j = 0; j < 4; ++j){
    const int cg = c0 + 16 * j + cl;
    const float brr = bih[cg]         + bhh[cg];
    const float bzz = bih[Hid + cg]   + bhh[Hid + cg];
    const float bin = bih[2*Hid + cg];
    const float bhn = bhh[2*Hid + cg];
#pragma unroll
    for (int v = 0; v < 8; ++v){
      const int row = m0 + rofs + v;
      const float r  = sigm(accR[j][v] + brr);
      const float z  = sigm(accZ[j][v] + bzz);
      const float nn = tanh_fast(accIN[j][v] + bin + r * (accHN[j][v] + bhn));
      const float hp = h_prev[(size_t)row * Hid + cg];
      const float hv = (1.0f - z) * nn + z * hp;
      h_new[(size_t)row * Hid + cg]    = hv;
      h_new_bf[(size_t)row * Hid + cg] = f32_to_bf16(hv);
    }
  }
}

// C = act(A[128xK] @ W[KxN] + bias); optional f32 (strided) and bf16 outputs.
__global__ __launch_bounds__(128) void mm_bias_act(
    const unsigned short* __restrict__ A, int lda, int K,
    const unsigned short* __restrict__ W, int Nn,
    const float* __restrict__ bias, int do_relu,
    float* __restrict__ outF, int ldoF,
    unsigned short* __restrict__ outB, int ldoB)
{
  const int lane = threadIdx.x & 31;
  const int wave = threadIdx.x >> 5;
  const int m0 = blockIdx.y * 64 + wave * 16;
  const int c0 = blockIdx.x * 64;

  v8f acc[4] = {};
  for (int k0 = 0; k0 < K; k0 += 32){
    v16bf a = load_a_frag(A, lda, m0, k0, lane);
#pragma unroll
    for (int j = 0; j < 4; ++j)
      acc[j] = wmma_bf16(a, load_b_frag(W, Nn, k0, c0 + 16*j, lane), acc[j]);
  }
  const int cl   = lane & 15;
  const int rofs = (lane >> 4) * 8;
#pragma unroll
  for (int j = 0; j < 4; ++j){
    const int cg = c0 + 16*j + cl;
    const float bv = bias[cg];
#pragma unroll
    for (int v = 0; v < 8; ++v){
      const int row = m0 + rofs + v;
      float val = acc[j][v] + bv;
      if (do_relu) val = fmaxf(val, 0.0f);
      if (outF) outF[(size_t)row * ldoF + cg] = val;
      if (outB) outB[(size_t)row * ldoB + cg] = f32_to_bf16(val);
    }
  }
}

// Transpose+convert weight [Nr, Kc] (row-major f32) -> K-major bf16 [Kc, Nr].
__global__ void pack_wt(const float* __restrict__ src, unsigned short* __restrict__ dst,
                        int Nr, int Kc){
  int idx = blockIdx.x * 256 + threadIdx.x;
  if (idx < Nr * Kc){
    int k = idx / Nr, n = idx - k * Nr;
    dst[idx] = f32_to_bf16(src[(size_t)n * Kc + k]);
  }
}
__global__ void cvt_bf16(const float* __restrict__ s, unsigned short* __restrict__ d, int n){
  int i = blockIdx.x * 256 + threadIdx.x;
  if (i < n) d[i] = f32_to_bf16(s[i]);
}
__global__ void fill_zero(unsigned int* __restrict__ p, int n){
  int i = blockIdx.x * 256 + threadIdx.x;
  if (i < n) p[i] = 0u;
}

extern "C" void kernel_launch(void* const* d_in, const int* in_sizes, int n_in,
                              void* d_out, int out_size, void* d_ws, size_t ws_size,
                              hipStream_t stream)
{
  const float* X     = (const float*)d_in[0];
  const float* eWih0 = (const float*)d_in[1];
  const float* eWhh0 = (const float*)d_in[2];
  const float* ebih0 = (const float*)d_in[3];
  const float* ebhh0 = (const float*)d_in[4];
  const float* eWih1 = (const float*)d_in[5];
  const float* eWhh1 = (const float*)d_in[6];
  const float* ebih1 = (const float*)d_in[7];
  const float* ebhh1 = (const float*)d_in[8];
  const float* dWih0 = (const float*)d_in[9];
  const float* dWhh0 = (const float*)d_in[10];
  const float* dbih0 = (const float*)d_in[11];
  const float* dbhh0 = (const float*)d_in[12];
  const float* dWih1 = (const float*)d_in[13];
  const float* dWhh1 = (const float*)d_in[14];
  const float* dbih1 = (const float*)d_in[15];
  const float* dbhh1 = (const float*)d_in[16];
  const float* projW = (const float*)d_in[17];
  const float* projb = (const float*)d_in[18];
  const float* outW  = (const float*)d_in[19];
  const float* outb  = (const float*)d_in[20];
  float* out = (float*)d_out;
  (void)in_sizes; (void)n_in; (void)out_size; (void)ws_size;

  char* wsb = (char*)d_ws;
  size_t off = 0;
  auto alloc = [&](size_t bytes) -> void* {
    void* p = wsb + off;
    off = (off + bytes + 255) & ~((size_t)255);
    return p;
  };

  unsigned short* pWih0e = (unsigned short*)alloc((size_t)Feat*3*Hid*2);
  unsigned short* pWhh0e = (unsigned short*)alloc((size_t)Hid*3*Hid*2);
  unsigned short* pWih1e = (unsigned short*)alloc((size_t)Hid*3*Hid*2);
  unsigned short* pWhh1e = (unsigned short*)alloc((size_t)Hid*3*Hid*2);
  unsigned short* pWih0d = (unsigned short*)alloc((size_t)Feat*3*Hid*2);
  unsigned short* pWhh0d = (unsigned short*)alloc((size_t)Hid*3*Hid*2);
  unsigned short* pWih1d = (unsigned short*)alloc((size_t)Hid*3*Hid*2);
  unsigned short* pWhh1d = (unsigned short*)alloc((size_t)Hid*3*Hid*2);
  unsigned short* pProj  = (unsigned short*)alloc((size_t)Hid*Prj*2);
  unsigned short* pOut   = (unsigned short*)alloc((size_t)Prj*Feat*2);
  unsigned short* Xbf    = (unsigned short*)alloc((size_t)Bsz*Tlen*Feat*2);
  float* h0f[2]; float* h1f[2]; unsigned short* h0b[2]; unsigned short* h1b[2];
  for (int i = 0; i < 2; ++i) h0f[i] = (float*)alloc((size_t)Bsz*Hid*4);
  for (int i = 0; i < 2; ++i) h1f[i] = (float*)alloc((size_t)Bsz*Hid*4);
  for (int i = 0; i < 2; ++i) h0b[i] = (unsigned short*)alloc((size_t)Bsz*Hid*2);
  for (int i = 0; i < 2; ++i) h1b[i] = (unsigned short*)alloc((size_t)Bsz*Hid*2);
  unsigned short* actb = (unsigned short*)alloc((size_t)Bsz*Prj*2);
  unsigned short* yb   = (unsigned short*)alloc((size_t)Bsz*Feat*2);

  auto packW = [&](const float* s, unsigned short* d, int Nr, int Kc){
    int total = Nr * Kc;
    pack_wt<<<(total + 255) / 256, 256, 0, stream>>>(s, d, Nr, Kc);
  };
  packW(eWih0, pWih0e, 3*Hid, Feat);
  packW(eWhh0, pWhh0e, 3*Hid, Hid);
  packW(eWih1, pWih1e, 3*Hid, Hid);
  packW(eWhh1, pWhh1e, 3*Hid, Hid);
  packW(dWih0, pWih0d, 3*Hid, Feat);
  packW(dWhh0, pWhh0d, 3*Hid, Hid);
  packW(dWih1, pWih1d, 3*Hid, Hid);
  packW(dWhh1, pWhh1d, 3*Hid, Hid);
  packW(projW, pProj, Prj, Hid);
  packW(outW,  pOut,  Feat, Prj);

  { int n = Bsz*Tlen*Feat; cvt_bf16<<<(n+255)/256, 256, 0, stream>>>(X, Xbf, n); }
  { int n = Bsz*Hid;   fill_zero<<<(n+255)/256, 256, 0, stream>>>((unsigned int*)h0f[0], n); }
  { int n = Bsz*Hid;   fill_zero<<<(n+255)/256, 256, 0, stream>>>((unsigned int*)h1f[0], n); }
  { int n = Bsz*Hid/2; fill_zero<<<(n+255)/256, 256, 0, stream>>>((unsigned int*)h0b[0], n); }
  { int n = Bsz*Hid/2; fill_zero<<<(n+255)/256, 256, 0, stream>>>((unsigned int*)h1b[0], n); }

  dim3 blk(128);
  dim3 grdStage(Hid/64, 2);
  int cur0 = 0, cur1 = 0;

  // ---- encoder ----
  for (int t = 0; t < Tlen; ++t){
    int n0 = cur0 ^ 1, n1 = cur1 ^ 1;
    gru_stage<<<grdStage, blk, 0, stream>>>(Xbf + (size_t)t*Feat, Tlen*Feat, Feat, pWih0e,
        h0b[cur0], pWhh0e, ebih0, ebhh0, h0f[cur0], h0f[n0], h0b[n0]);
    gru_stage<<<grdStage, blk, 0, stream>>>(h0b[n0], Hid, Hid, pWih1e,
        h1b[cur1], pWhh1e, ebih1, ebhh1, h1f[cur1], h1f[n1], h1b[n1]);
    cur0 = n0; cur1 = n1;
  }

  // ---- decoder (h carries over from encoder) ----
  const unsigned short* curIn = Xbf + (size_t)(Tlen - 1) * Feat;
  int ldIn = Tlen * Feat;
  dim3 grdProj(Prj/64, 2);
  dim3 grdOut(Feat/64, 2);
  for (int s = 0; s < OUTL; ++s){
    int n0 = cur0 ^ 1, n1 = cur1 ^ 1;
    gru_stage<<<grdStage, blk, 0, stream>>>(curIn, ldIn, Feat, pWih0d,
        h0b[cur0], pWhh0d, dbih0, dbhh0, h0f[cur0], h0f[n0], h0b[n0]);
    gru_stage<<<grdStage, blk, 0, stream>>>(h0b[n0], Hid, Hid, pWih1d,
        h1b[cur1], pWhh1d, dbih1, dbhh1, h1f[cur1], h1f[n1], h1b[n1]);
    mm_bias_act<<<grdProj, blk, 0, stream>>>(h1b[n1], Hid, Hid, pProj, Prj,
        projb, 1, (float*)nullptr, 0, actb, Prj);
    mm_bias_act<<<grdOut, blk, 0, stream>>>(actb, Prj, Prj, pOut, Feat,
        outb, 0, out + (size_t)s*Feat, OUTL*Feat, yb, Feat);
    curIn = yb; ldIn = Feat;
    cur0 = n0; cur1 = n1;
  }
}
